// SNNCore_87909390614627
// MI455X (gfx1250) — matrix operand order repaired and checked by reading, not compiled, not run on hardware
//
#include <hip/hip_runtime.h>
#include <hip/hip_bf16.h>

typedef __attribute__((ext_vector_type(8)))  float  v8f;
typedef __attribute__((ext_vector_type(16))) __bf16 v16bf;
typedef __attribute__((ext_vector_type(8)))  __bf16 v8bf;

#define D_   1024
#define L_   4
#define S_   512
#define T_   4
#define B_   4
#define V_   32000
#define H_   (L_ * D_)   // 4096
#define M_   (B_ * S_)   // 2048
#define NBLK 16          // persistent workgroups for the recurrent core

// ---------------------------------------------------------------------------
// WMMA fragment helpers (wave32; A/B in contiguous k order k = 16*(lane/16) + s)
// ---------------------------------------------------------------------------
__device__ __forceinline__ v16bf frag_from2(v8bf lo, v8bf hi) {
  v16bf r;
#pragma unroll
  for (int i = 0; i < 8; ++i) { r[i] = lo[i]; r[i + 8] = hi[i]; }
  return r;
}

// A fragment 16x32 bf16 from row-major buffer (row stride in elements).
// lane<16 : row=lane,   K = kbase+[0..15]
// lane>=16: row=lane-16,K = kbase+[16..31]
__device__ __forceinline__ v16bf load_a_frag(const __bf16* base, int stride,
                                             int lane, int kbase) {
  const int m    = lane & 15;
  const int koff = kbase + ((lane >> 4) << 4);
  const __bf16* p = base + m * stride + koff;
  return frag_from2(*(const v8bf*)p, *(const v8bf*)(p + 8));
}

// B fragment 32x16 bf16: lane holds column n=lane&15, 16 contiguous K values
// (hi half-wave takes K+16). p must point at W[n][kbase + (lane>=16 ? 16:0)].
__device__ __forceinline__ v16bf load_b_frag(const __bf16* p) {
  return frag_from2(*(const v8bf*)p, *(const v8bf*)(p + 8));
}

__device__ __forceinline__ v8f wmma_bf16(v16bf a, v16bf b, v8f c) {
  return __builtin_amdgcn_wmma_f32_16x16x32_bf16(false, a, false, b,
                                                 (short)0, c, false, false);
}

__device__ __forceinline__ v8bf pack8(float4 a, float4 b) {
  v8bf v;
  v[0] = (__bf16)a.x; v[1] = (__bf16)a.y; v[2] = (__bf16)a.z; v[3] = (__bf16)a.w;
  v[4] = (__bf16)b.x; v[5] = (__bf16)b.y; v[6] = (__bf16)b.z; v[7] = (__bf16)b.w;
  return v;
}

// 32-element row staged through registers (f32 or bf16 source)
struct RowF32 { float4 q[8]; };
struct RowB16 { uint4  q[4]; };

__device__ __forceinline__ RowF32 gload_row(const float* p) {
  RowF32 r; const float4* s = (const float4*)p;
#pragma unroll
  for (int c = 0; c < 8; ++c) r.q[c] = s[c];
  return r;
}
__device__ __forceinline__ RowB16 gload_row(const __bf16* p) {
  RowB16 r; const uint4* s = (const uint4*)p;
#pragma unroll
  for (int c = 0; c < 4; ++c) r.q[c] = s[c];
  return r;
}
__device__ __forceinline__ void lds_store_row(__bf16* dst, const RowF32& r) {
#pragma unroll
  for (int c = 0; c < 4; ++c)
    *(v8bf*)&dst[c * 8] = pack8(r.q[2 * c], r.q[2 * c + 1]);
}
__device__ __forceinline__ void lds_store_row(__bf16* dst, const RowB16& r) {
#pragma unroll
  for (int c = 0; c < 4; ++c)
    *(uint4*)&dst[c * 8] = r.q[c];
}

// ---------------------------------------------------------------------------
// Device-scope barrier for the persistent recurrent kernel
// ---------------------------------------------------------------------------
__device__ __forceinline__ void global_barrier(unsigned* cnt, unsigned* rel,
                                               unsigned phase) {
  __threadfence();
  __syncthreads();
  if (threadIdx.x == 0) {
    unsigned prev = __hip_atomic_fetch_add(cnt, 1u, __ATOMIC_ACQ_REL,
                                           __HIP_MEMORY_SCOPE_AGENT);
    if (prev == (unsigned)(NBLK - 1)) {
      __hip_atomic_store(cnt, 0u, __ATOMIC_RELAXED, __HIP_MEMORY_SCOPE_AGENT);
      __hip_atomic_store(rel, phase, __ATOMIC_RELEASE, __HIP_MEMORY_SCOPE_AGENT);
    } else {
      while (__hip_atomic_load(rel, __ATOMIC_ACQUIRE,
                               __HIP_MEMORY_SCOPE_AGENT) < phase) {
        __builtin_amdgcn_s_sleep(2);
      }
    }
  }
  __syncthreads();
  __threadfence();
}

// ---------------------------------------------------------------------------
// Small utility kernels
// ---------------------------------------------------------------------------
__global__ void zero_kernel(unsigned* p, int n) {
  for (int i = blockIdx.x * blockDim.x + threadIdx.x; i < n;
       i += gridDim.x * blockDim.x)
    p[i] = 0u;
}

__global__ void cvt_bf16_kernel(const float* __restrict__ in,
                                __bf16* __restrict__ out, int n) {
  for (int i = blockIdx.x * blockDim.x + threadIdx.x; i < n;
       i += gridDim.x * blockDim.x)
    out[i] = (__bf16)in[i];
}

// ---------------------------------------------------------------------------
// Generic tiled GEMM:  C[m][n] = sum_k A[g(m)][k] * Bw[n][k] + bias[n]
// A (f32 or bf16, optional row gather) and Bw (f32) are converted/staged to
// bf16 LDS, f32 WMMA accumulate. BM=512, BN=128, 512 threads (16 waves as
// 8(M) x 2(N)); each wave owns 4x4 16x16 tiles. Software-pipelined: global
// loads for step i+1 issue before compute of step i; ping-pong LDS gives a
// single __syncthreads per K-step. Requires M%512==0, N%128==0, K%32==0.
// ---------------------------------------------------------------------------
#define BM  512
#define BN  128
#define LDT 40   // LDS row stride (elements) to de-conflict banks

template <typename AT>
__global__ __launch_bounds__(512) void gemm_bf16_kernel(
    const AT* __restrict__ Araw, const int* __restrict__ gidx, int K,
    const float* __restrict__ Bw, const float* __restrict__ bias,
    float* __restrict__ C, int N) {
  __shared__ __bf16 As[2 * BM * LDT];
  __shared__ __bf16 Bs[2 * BN * LDT];
  const int tid  = threadIdx.x;
  const int lane = tid & 31;
  const int wave = tid >> 5;          // 0..15
  const int wm = wave & 7, wn = wave >> 3;
  const int lm = lane & 15, lh = lane >> 4;
  const int bm = blockIdx.y * BM, bn = blockIdx.x * BN;

  v8f acc[4][4];
#pragma unroll
  for (int ni = 0; ni < 4; ++ni) {
    const float bv = bias[bn + wn * 64 + ni * 16 + lm];
#pragma unroll
    for (int mi = 0; mi < 4; ++mi)
#pragma unroll
      for (int r = 0; r < 8; ++r) acc[mi][ni][r] = bv;
  }

  const int arow = bm + tid;                 // one A row per thread
  const int agr  = gidx ? gidx[arow] : arow; // optional row gather
  const AT* asrc = Araw + (size_t)agr * K;
  const float* bsrc =
      Bw + (size_t)(bn + (tid >> 2)) * K + (tid & 3) * 8;  // 8 f32 per thread

  auto   areg = gload_row(asrc);
  float4 b0 = ((const float4*)bsrc)[0];
  float4 b1 = ((const float4*)bsrc)[1];
  int buf = 0;

  for (int k0 = 0; k0 < K; k0 += 32) {
    __bf16* Asb = As + buf * (BM * LDT);
    __bf16* Bsb = Bs + buf * (BN * LDT);
    lds_store_row(&Asb[tid * LDT], areg);
    *(v8bf*)&Bsb[(tid >> 2) * LDT + (tid & 3) * 8] = pack8(b0, b1);
    __syncthreads();

    if (k0 + 32 < K) {  // issue next step's global loads before computing
      areg = gload_row(asrc + k0 + 32);
      b0 = ((const float4*)(bsrc + k0 + 32))[0];
      b1 = ((const float4*)(bsrc + k0 + 32))[1];
      __builtin_prefetch(bsrc + k0 + 64, 0, 1);
    }

    v16bf bf[4];
#pragma unroll
    for (int ni = 0; ni < 4; ++ni)
      bf[ni] = load_b_frag(&Bsb[(wn * 64 + ni * 16 + lm) * LDT + lh * 16]);
#pragma unroll
    for (int mi = 0; mi < 4; ++mi) {
      v16bf af = load_a_frag(&Asb[(wm * 64 + mi * 16) * LDT], LDT, lane, 0);
#pragma unroll
      for (int ni = 0; ni < 4; ++ni)
        acc[mi][ni] = wmma_bf16(af, bf[ni], acc[mi][ni]);
    }
    buf ^= 1;
  }

#pragma unroll
  for (int mi = 0; mi < 4; ++mi)
#pragma unroll
    for (int ni = 0; ni < 4; ++ni) {
      const int col  = bn + wn * 64 + ni * 16 + lm;
      const int rowb = bm + wm * 64 + mi * 16 + lh * 8;  // C: VGPR r -> M=r(+8)
#pragma unroll
      for (int r = 0; r < 8; ++r)
        C[(size_t)(rowb + r) * N + col] = acc[mi][ni][r];
    }
}

// ---------------------------------------------------------------------------
// Recurrent persistent kernel: 16 WGs x 128 threads (4 waves). Wave owns one
// 16-column N-tile (64 waves == 64 tiles of D=1024). One device barrier per
// layer (err ping-pong makes a second barrier unnecessary).
// ---------------------------------------------------------------------------
__device__ __forceinline__ void ln_stage_row(const float* __restrict__ row,
                                             const float* __restrict__ g,
                                             const float* __restrict__ b,
                                             __bf16* __restrict__ dst,
                                             int lane) {
  float4 v[8];
  const float4* s4 = (const float4*)(row + lane * 32);
#pragma unroll
  for (int c = 0; c < 8; ++c) v[c] = s4[c];
  float sum = 0.f, sq = 0.f;
#pragma unroll
  for (int c = 0; c < 8; ++c) {
    sum += v[c].x + v[c].y + v[c].z + v[c].w;
    sq  += v[c].x * v[c].x + v[c].y * v[c].y + v[c].z * v[c].z + v[c].w * v[c].w;
  }
#pragma unroll
  for (int off = 16; off > 0; off >>= 1) {
    sum += __shfl_xor(sum, off, 32);
    sq  += __shfl_xor(sq, off, 32);
  }
  const float mu = sum * (1.f / 1024.f);
  const float rs = rsqrtf(sq * (1.f / 1024.f) - mu * mu + 1e-5f);
  const float4* g4 = (const float4*)(g + lane * 32);
  const float4* b4 = (const float4*)(b + lane * 32);
#pragma unroll
  for (int c = 0; c < 4; ++c) {
    float4 xa = v[2 * c], xb = v[2 * c + 1];
    float4 ga = g4[2 * c], gb = g4[2 * c + 1];
    float4 ba = b4[2 * c], bb = b4[2 * c + 1];
    v8bf o;
    o[0] = (__bf16)((xa.x - mu) * rs * ga.x + ba.x);
    o[1] = (__bf16)((xa.y - mu) * rs * ga.y + ba.y);
    o[2] = (__bf16)((xa.z - mu) * rs * ga.z + ba.z);
    o[3] = (__bf16)((xa.w - mu) * rs * ga.w + ba.w);
    o[4] = (__bf16)((xb.x - mu) * rs * gb.x + bb.x);
    o[5] = (__bf16)((xb.y - mu) * rs * gb.y + bb.y);
    o[6] = (__bf16)((xb.z - mu) * rs * gb.z + bb.z);
    o[7] = (__bf16)((xb.w - mu) * rs * gb.w + bb.w);
    *(v8bf*)&dst[lane * 32 + c * 8] = o;
  }
}

__global__ __launch_bounds__(128) void recurrent_kernel(
    const float* __restrict__ xbsd,                       // [B][S][D]
    const __bf16* __restrict__ genW, const float* __restrict__ genB,
    const __bf16* __restrict__ infW, const float* __restrict__ infB,
    const float* __restrict__ nsg, const float* __restrict__ nsb,
    const float* __restrict__ neg_, const float* __restrict__ neb,
    const float* __restrict__ escale,
    float* __restrict__ states,                           // [L][16][D], rows>=4 zero
    float* __restrict__ errbuf,                           // [2][4][D]
    __bf16* __restrict__ hid,                             // [M][H] bf16
    unsigned* __restrict__ barcnt, unsigned* __restrict__ barrel) {
  __shared__ __bf16 A[16 * 1032];                         // padded stride 1032
  const int tid  = threadIdx.x;
  const int lane = tid & 31;
  const int wave = tid >> 5;                              // 0..3 (also LN row)
  const int lm = lane & 15, lh = lane >> 4;
  const int tile = blockIdx.x * 4 + wave;                 // 0..63 N-tile
  const int myn  = tile * 16 + lm;                        // owned column
  unsigned phase = 0;

  for (int i = tid; i < 12 * 1032; i += 128) A[4 * 1032 + i] = (__bf16)0.0f;
  __syncthreads();

  for (int t = 0; t < T_; ++t) {
    const bool last = (t == T_ - 1);
    for (int s = 0; s < S_; ++s) {
      for (int j = 0; j < L_; ++j) {
        // ---- A_s = LayerNorm(states[j]) rows 0..3 ----
        ln_stage_row(states + (size_t)(j * 16 + wave) * D_,
                     nsg + j * D_, nsb + j * D_, A + wave * 1032, lane);
        __syncthreads();

        // ---- GEMM1: pred_pre = A_s @ gen_W[j]^T + gen_b[j] ----
        v8f acc;
        {
          const float bv = genB[j * D_ + myn];
#pragma unroll
          for (int r = 0; r < 8; ++r) acc[r] = bv;
        }
        {
          const __bf16* Wrow =
              genW + (size_t)j * D_ * D_ + (size_t)myn * D_ + lh * 16;
          for (int ks = 0; ks < 32; ++ks) {
            v16bf af = load_a_frag(A, 1032, lane, ks * 32);
            v16bf bf = load_b_frag(Wrow + ks * 32);
            acc = wmma_bf16(af, bf, acc);
          }
        }

        // ---- spike, err (column owner, rows 0..3) ----
        const float esc  = escale[j];
        float* eout      = errbuf + (size_t)(j & 1) * 4 * D_;
        const float* ein = errbuf + (size_t)((j ^ 1) & 1) * 4 * D_;
        if (lh == 0) {
#pragma unroll
          for (int r = 0; r < 4; ++r) {
            const float pred = (acc[r] * 0.5f - 1.0f > 0.0f) ? 1.0f : 0.0f;
            const float bu = (j == 0)
                                 ? xbsd[((size_t)r * S_ + s) * D_ + myn]
                                 : ein[r * D_ + myn];
            eout[r * D_ + myn] = (bu - pred) * esc;
          }
        }
        ++phase;
        global_barrier(barcnt, barrel, phase);

        // ---- A_e = LayerNorm(err) rows 0..3 ----
        ln_stage_row(eout + (size_t)wave * D_, neg_ + j * D_, neb + j * D_,
                     A + wave * 1032, lane);
        __syncthreads();

        // ---- GEMM2: upd_pre = A_e @ inf_W[j]^T + inf_b[j] ----
        {
          const float bv = infB[j * D_ + myn];
#pragma unroll
          for (int r = 0; r < 8; ++r) acc[r] = bv;
        }
        {
          const __bf16* Wrow =
              infW + (size_t)j * D_ * D_ + (size_t)myn * D_ + lh * 16;
          for (int ks = 0; ks < 32; ++ks) {
            v16bf af = load_a_frag(A, 1032, lane, ks * 32);
            v16bf bf = load_b_frag(Wrow + ks * 32);
            acc = wmma_bf16(af, bf, acc);
          }
        }

        // ---- state EMA + (last pass) hid collection (bf16) ----
        if (lh == 0) {
#pragma unroll
          for (int r = 0; r < 4; ++r) {
            const float upd = (acc[r] * 0.5f - 1.0f > 0.0f) ? 1.0f : 0.0f;
            float* sp = states + (size_t)(j * 16 + r) * D_ + myn;
            const float nsv = sp[0] * 0.9f + upd * 0.1f;
            sp[0] = nsv;
            if (last)
              hid[((size_t)r * S_ + s) * H_ + j * D_ + myn] = (__bf16)nsv;
          }
        }
      }
    }
  }
}

// ---------------------------------------------------------------------------
// Host launcher
// ---------------------------------------------------------------------------
extern "C" void kernel_launch(void* const* d_in, const int* in_sizes, int n_in,
                              void* d_out, int out_size, void* d_ws,
                              size_t ws_size, hipStream_t stream) {
  (void)in_sizes; (void)n_in; (void)out_size; (void)ws_size;
  const int*   ids    = (const int*)d_in[0];
  const float* emb    = (const float*)d_in[1];
  const float* encW   = (const float*)d_in[2];
  const float* encB   = (const float*)d_in[3];
  const float* genW   = (const float*)d_in[4];
  const float* genB   = (const float*)d_in[5];
  const float* infW   = (const float*)d_in[6];
  const float* infB   = (const float*)d_in[7];
  const float* nsg    = (const float*)d_in[8];
  const float* nsb    = (const float*)d_in[9];
  const float* neg_   = (const float*)d_in[10];
  const float* neb    = (const float*)d_in[11];
  const float* escale = (const float*)d_in[12];
  const float* outW   = (const float*)d_in[13];
  const float* outB   = (const float*)d_in[14];
  float* logits = (float*)d_out;

  char* ws = (char*)d_ws;
  size_t off = 0;
  auto take = [&](size_t bytes) -> char* {
    char* p = ws + off;
    off = (off + bytes + 255) & ~(size_t)255;
    return p;
  };
  unsigned* bar    = (unsigned*)take(256);
  float*    states = (float*)take((size_t)L_ * 16 * D_ * 4);
  float*    errbuf = (float*)take((size_t)2 * 4 * D_ * 4);
  size_t zero_bytes = off;                 // bar + states + errbuf
  float*    xbsd   = (float*)take((size_t)M_ * D_ * 4);
  __bf16*   hid    = (__bf16*)take((size_t)M_ * H_ * 2);
  __bf16*   genWb  = (__bf16*)take((size_t)L_ * D_ * D_ * 2);
  __bf16*   infWb  = (__bf16*)take((size_t)L_ * D_ * D_ * 2);

  // 1) init barrier + states + err buffers
  const int zwords = (int)(zero_bytes / 4);
  zero_kernel<<<(zwords + 255) / 256, 256, 0, stream>>>((unsigned*)ws, zwords);

  // 2) pre-convert recurrent weights to bf16 (halves L2 traffic x2048 reuse)
  const int wn = L_ * D_ * D_;
  cvt_bf16_kernel<<<2048, 256, 0, stream>>>(genW, genWb, wn);
  cvt_bf16_kernel<<<2048, 256, 0, stream>>>(infW, infWb, wn);

  // 3) encoder: x[b][s][e] = emb[ids[b,s]] @ enc_W^T + enc_b
  gemm_bf16_kernel<float><<<dim3(D_ / BN, M_ / BM), 512, 0, stream>>>(
      emb, ids, D_, encW, encB, xbsd, D_);

  // 4) recurrent core (persistent, device-barrier synced)
  recurrent_kernel<<<NBLK, 128, 0, stream>>>(
      xbsd, genWb, genB, infWb, infB, nsg, nsb, neg_, neb, escale, states,
      errbuf, hid, bar, bar + 1);

  // 5) output projection: logits = hid @ out_W^T + out_b  (A is bf16)
  gemm_bf16_kernel<__bf16><<<dim3(V_ / BN, M_ / BM), 512, 0, stream>>>(
      hid, nullptr, H_, outW, outB, logits, V_);
}
